// SalT_4544075399566
// MI455X (gfx1250) — matrix operand order repaired and checked by reading, not compiled, not run on hardware
//
#include <hip/hip_runtime.h>
#include <hip/hip_bf16.h>

typedef __attribute__((ext_vector_type(16))) __bf16 v16bf;
typedef __attribute__((ext_vector_type(8)))  float  v8f;
typedef __attribute__((ext_vector_type(4)))  float  v4f;

#define NREL   16
#define NROWS  4096
#define DIM    64
#define HID    256

// ---------------------------------------------------------------------------
// Deterministic Fisher-Yates reset mask (20% of rows -> 0).  Single wave.
// ---------------------------------------------------------------------------
__global__ void mask_kernel(float* __restrict__ mask) {
  __shared__ int perm[NROWS];
  if (threadIdx.x == 0) {
    for (int i = 0; i < NROWS; ++i) perm[i] = i;
    unsigned long long s = 0x9E3779B97F4A7C15ull;
    for (int i = NROWS - 1; i > 0; --i) {
      s += 0x9E3779B97F4A7C15ull;
      unsigned long long z = s;
      z = (z ^ (z >> 30)) * 0xBF58476D1CE4E5B9ull;
      z = (z ^ (z >> 27)) * 0x94D049BB133111EBull;
      z ^= z >> 31;
      int j = (int)(z % (unsigned long long)(i + 1));
      int t = perm[i]; perm[i] = perm[j]; perm[j] = t;
    }
    for (int i = 0; i < NROWS; ++i) mask[i] = 1.0f;
    for (int i = 0; i < NROWS / 5; ++i) mask[perm[i]] = 0.0f;  // 819 ~= int(4096*0.2)
  }
}

// ---------------------------------------------------------------------------
// L1-normalize, ent = x@Win, gather prev, gate, h = g*ent + (1-g)*prev
// block: 256 thr = 4 rows x 64 dims
// ---------------------------------------------------------------------------
__global__ __launch_bounds__(256) void prep_kernel(
    const float* __restrict__ feat, const int* __restrict__ idxs,
    const float* __restrict__ mem,  const float* __restrict__ Win,
    const float* __restrict__ Wg,   const float* __restrict__ bg,
    const float* __restrict__ mask, float* __restrict__ h) {
  __shared__ float ent_s[4][DIM];
  __shared__ float prev_s[4][DIM];
  const int lr  = threadIdx.x >> 6;
  const int d   = threadIdx.x & 63;
  const int row = blockIdx.x * 4 + lr;

  const float* xr = feat + row * 32;
  float l1 = 0.f;
#pragma unroll
  for (int j = 0; j < 32; ++j) l1 += fabsf(xr[j]);
  const float inv = 1.0f / fmaxf(l1, 1e-12f);
  float e = 0.f;
#pragma unroll
  for (int j = 0; j < 32; ++j) e += xr[j] * inv * Win[j * DIM + d];
  const float p = mem[(size_t)idxs[row] * DIM + d] * mask[row];
  ent_s[lr][d]  = e;
  prev_s[lr][d] = p;
  __syncthreads();
  float acc = bg[d];
#pragma unroll 8
  for (int j = 0; j < DIM; ++j) acc += ent_s[lr][j]  * Wg[j * DIM + d];
#pragma unroll 8
  for (int j = 0; j < DIM; ++j) acc += prev_s[lr][j] * Wg[(DIM + j) * DIM + d];
  const float g = 1.0f / (1.0f + __expf(-acc));
  h[row * DIM + d] = g * e + (1.0f - g) * p;
}

// ---------------------------------------------------------------------------
// Wqk[l] = Wq[l] @ Wk[l]^T   (64x64 per layer; folds q/k projections)
// ---------------------------------------------------------------------------
__global__ void wqk_kernel(const float* __restrict__ Wq, const float* __restrict__ Wk,
                           float* __restrict__ Wqk) {
  const int t = blockIdx.x * blockDim.x + threadIdx.x;
  if (t >= 2 * DIM * DIM) return;
  const int l = t >> 12, dp = (t >> 6) & 63, d = t & 63;
  const float* q = Wq + (l * DIM + dp) * 32;
  const float* k = Wk + (l * DIM + d) * 32;
  float s = 0.f;
#pragma unroll
  for (int a = 0; a < 32; ++a) s += q[a] * k[a];
  Wqk[t] = s;
}

// ---------------------------------------------------------------------------
// Pre-swizzle h (fp32 [4096,64]) into WMMA B-fragment order (bf16):
// hB[kblk(128)][ntile(4)][lane(32)][e(16)], lane: half=lane/16, n=nt*16+lane%16,
// element e -> K = kblk*32 + half*8 + e (e<8), kblk*32 + 16 + half*8 + (e-8) (e>=8)   (16-bit B layout, wave32)
// ---------------------------------------------------------------------------
__global__ void make_hB_kernel(const float* __restrict__ h, __bf16* __restrict__ hB) {
  const int t = blockIdx.x * blockDim.x + threadIdx.x;
  if (t >= 128 * 4 * 32) return;
  const int lane = t & 31, nt = (t >> 5) & 3, kblk = t >> 7;
  const int half = lane >> 4;
  const int n    = nt * 16 + (lane & 15);
  const int kb   = kblk * 32 + half * 8;
  __bf16* dst = hB + (size_t)t * 16;
#pragma unroll
  for (int e = 0; e < 16; ++e) dst[e] = (__bf16)h[(kb + e + (e >= 8 ? 8 : 0)) * DIM + n];
}

// ---------------------------------------------------------------------------
// m[r, nblk*16 .. +15, 0:64] = adj[r, rows, :] (fp32 -> bf16 in-reg) @ hB
// 1 wave per (r, nblk) task; 8 waves / block; 128 K-iters x 4 WMMA.
// ---------------------------------------------------------------------------
__global__ __launch_bounds__(256) void spmm_kernel(
    const float* __restrict__ adj, const __bf16* __restrict__ hB,
    float* __restrict__ m_out) {
  const int tid  = threadIdx.x;
  const int lane = tid & 31;
  const int wave = tid >> 5;
  const int task = blockIdx.x * 8 + wave;   // 0..4095
  const int r    = task >> 8;               // 0..15
  const int nblk = task & 255;              // 0..255
  const int half = lane >> 4;
  const int mrow = lane & 15;

  // A-fragment base: lane holds row (nblk*16+mrow); K chunks {half*8+0..7} and {16+half*8+0..7}
  const float* arow = adj + ((size_t)(r * NROWS + nblk * 16 + mrow)) * NROWS + half * 8;
  const v16bf* bbase = (const v16bf*)hB + lane;   // + kblk*128 + nt*32

  v8f acc0 = {}, acc1 = {}, acc2 = {}, acc3 = {};

  for (int kk = 0; kk < NROWS; kk += 32) {
    const v4f a0 = __builtin_nontemporal_load((const v4f*)(arow + kk));
    const v4f a1 = __builtin_nontemporal_load((const v4f*)(arow + kk + 4));
    const v4f a2 = __builtin_nontemporal_load((const v4f*)(arow + kk + 16));
    const v4f a3 = __builtin_nontemporal_load((const v4f*)(arow + kk + 20));
    v16bf av;
    av[0]  = (__bf16)a0[0]; av[1]  = (__bf16)a0[1]; av[2]  = (__bf16)a0[2]; av[3]  = (__bf16)a0[3];
    av[4]  = (__bf16)a1[0]; av[5]  = (__bf16)a1[1]; av[6]  = (__bf16)a1[2]; av[7]  = (__bf16)a1[3];
    av[8]  = (__bf16)a2[0]; av[9]  = (__bf16)a2[1]; av[10] = (__bf16)a2[2]; av[11] = (__bf16)a2[3];
    av[12] = (__bf16)a3[0]; av[13] = (__bf16)a3[1]; av[14] = (__bf16)a3[2]; av[15] = (__bf16)a3[3];

    const v16bf* bp = bbase + (size_t)(kk >> 5) * 128;
    const v16bf b0 = bp[0];
    const v16bf b1 = bp[32];
    const v16bf b2 = bp[64];
    const v16bf b3 = bp[96];

    acc0 = __builtin_amdgcn_wmma_f32_16x16x32_bf16(false, av, false, b0, (short)0, acc0, false, false);
    acc1 = __builtin_amdgcn_wmma_f32_16x16x32_bf16(false, av, false, b1, (short)0, acc1, false, false);
    acc2 = __builtin_amdgcn_wmma_f32_16x16x32_bf16(false, av, false, b2, (short)0, acc2, false, false);
    acc3 = __builtin_amdgcn_wmma_f32_16x16x32_bf16(false, av, false, b3, (short)0, acc3, false, false);
  }

  // C/D layout: lane L, vgpr v -> row M = v + 8*half, col N = nt*16 + (L%16)
  float* mo = m_out + ((size_t)r * NROWS + nblk * 16) * DIM;
#pragma unroll
  for (int v = 0; v < 8; ++v) {
    const int rowoff = (v + 8 * half) * DIM + mrow;
    mo[rowoff +  0] = acc0[v];
    mo[rowoff + 16] = acc1[v];
    mo[rowoff + 32] = acc2[v];
    mo[rowoff + 48] = acc3[v];
  }
}

// ---------------------------------------------------------------------------
// Attention over relations + FFN + residual, fused per 16-row block.
// score[r,n] = (m[r,n,:] . (h[n,:]@Wqk)) / sqrt(32); softmax over r;
// h_out = h + relu(agg@W1)@W2
// ---------------------------------------------------------------------------
__global__ __launch_bounds__(256) void attn_ffn_kernel(
    const float* __restrict__ h_in, const float* __restrict__ m,
    const float* __restrict__ Wqk,  const float* __restrict__ W1,
    const float* __restrict__ W2,   float* __restrict__ h_out) {
  __shared__ float h_s[16][DIM];
  __shared__ float u_s[16][DIM];
  __shared__ float sc_s[16][NREL];
  __shared__ float agg_s[16][DIM];
  __shared__ float hid_s[16][HID];
  const int tid  = threadIdx.x;
  const int row0 = blockIdx.x * 16;

  for (int i = tid; i < 16 * DIM; i += 256) h_s[i >> 6][i & 63] = h_in[row0 * DIM + i];
  __syncthreads();

  for (int i = tid; i < 16 * DIM; i += 256) {
    const int rr = i >> 6, d = i & 63;
    float s = 0.f;
#pragma unroll 8
    for (int j = 0; j < DIM; ++j) s += h_s[rr][j] * Wqk[j * DIM + d];
    u_s[rr][d] = s;
  }
  __syncthreads();

  {
    const int rr = tid >> 4, r = tid & 15;
    const float* mp = m + ((size_t)r * NROWS + row0 + rr) * DIM;
    float s = 0.f;
#pragma unroll 8
    for (int d = 0; d < DIM; ++d) s += mp[d] * u_s[rr][d];
    sc_s[rr][r] = s * 0.17677669529663687f;  // 1/sqrt(ATT_DIM)
  }
  __syncthreads();

  if (tid < 16) {
    float mx = -1e30f;
#pragma unroll
    for (int r = 0; r < NREL; ++r) mx = fmaxf(mx, sc_s[tid][r]);
    float sum = 0.f;
#pragma unroll
    for (int r = 0; r < NREL; ++r) { const float e = __expf(sc_s[tid][r] - mx); sc_s[tid][r] = e; sum += e; }
    const float inv = 1.0f / sum;
#pragma unroll
    for (int r = 0; r < NREL; ++r) sc_s[tid][r] *= inv;
  }
  __syncthreads();

  for (int i = tid; i < 16 * DIM; i += 256) {
    const int rr = i >> 6, d = i & 63;
    float s = 0.f;
#pragma unroll
    for (int r = 0; r < NREL; ++r) s += sc_s[rr][r] * m[((size_t)r * NROWS + row0 + rr) * DIM + d];
    agg_s[rr][d] = s;
  }
  __syncthreads();

  for (int i = tid; i < 16 * HID; i += 256) {
    const int rr = i >> 8, j = i & 255;
    float s = 0.f;
#pragma unroll 8
    for (int d = 0; d < DIM; ++d) s += agg_s[rr][d] * W1[d * HID + j];
    hid_s[rr][j] = fmaxf(s, 0.0f);
  }
  __syncthreads();

  for (int i = tid; i < 16 * DIM; i += 256) {
    const int rr = i >> 6, d = i & 63;
    float s = 0.f;
#pragma unroll 8
    for (int j = 0; j < HID; ++j) s += hid_s[rr][j] * W2[j * DIM + d];
    h_out[(row0 + rr) * DIM + d] = h_s[rr][d] + s;
  }
}

// ---------------------------------------------------------------------------
extern "C" void kernel_launch(void* const* d_in, const int* in_sizes, int n_in,
                              void* d_out, int out_size, void* d_ws, size_t ws_size,
                              hipStream_t stream) {
  const float* feat = (const float*)d_in[0];
  const int*   idxs = (const int*)  d_in[1];
  const float* adj  = (const float*)d_in[2];
  const float* mem  = (const float*)d_in[3];
  const float* Win  = (const float*)d_in[4];
  const float* Wg   = (const float*)d_in[5];
  const float* bg   = (const float*)d_in[6];
  const float* Wq   = (const float*)d_in[7];
  const float* Wk   = (const float*)d_in[8];
  const float* W1   = (const float*)d_in[9];
  const float* W2   = (const float*)d_in[10];

  char*   ws      = (char*)d_ws;
  float*  ws_h    = (float*)(ws);                                   // 1 MiB
  __bf16* ws_hB   = (__bf16*)(ws + (1u << 20));                     // 512 KiB
  float*  ws_m    = (float*)(ws + (1u << 20) + (1u << 19));         // 16 MiB
  float*  ws_mask = (float*)(ws + (1u << 20) + (1u << 19) + (16u << 20));
  float*  ws_wqk  = ws_mask + NROWS;                                // 2*64*64

  mask_kernel<<<1, 32, 0, stream>>>(ws_mask);
  prep_kernel<<<NROWS / 4, 256, 0, stream>>>(feat, idxs, mem, Win, Wg, bg, ws_mask, ws_h);
  wqk_kernel<<<32, 256, 0, stream>>>(Wq, Wk, ws_wqk);

  for (int l = 0; l < 2; ++l) {
    make_hB_kernel<<<64, 256, 0, stream>>>(ws_h, ws_hB);
    spmm_kernel<<<512, 256, 0, stream>>>(adj, ws_hB, ws_m);
    float* hout = (l == 1) ? (float*)d_out : ws_h;
    attn_ffn_kernel<<<NROWS / 16, 256, 0, stream>>>(
        ws_h, ws_m, ws_wqk + l * DIM * DIM, W1 + l * DIM * HID, W2 + l * HID * DIM, hout);
  }
}
